// LSTM_76682346103213
// MI455X (gfx1250) — matrix operand order repaired and checked by reading, not compiled, not run on hardware
//
#include <hip/hip_runtime.h>
#include <math.h>

// ------------------------------------------------------------------
// LSTM forward for MI455X (gfx1250, wave32, WMMA bf16 16x16x32)
//   N=64 batch, T=512 steps, D=1024 in, H=1024 hidden
// Phase 0: pack Wx/Wh into WMMA B-fragment layout (bf16), cvt x->bf16,
//          init h buffer from h0, zero grid barrier.
// Phase 1: xW[T][N][4096] = x @ Wx + b  (B staged in LDS; software
//          pipeline with 2-stage manual unroll -> no register copies)
// Phase 2: persistent recurrent kernel; Wh fragments staged in LDS ONCE,
//          2-stage pipelined K-loop, cell state in registers, h ping-pong
//          bf16 buffer in L2, grid barrier per step.
// ------------------------------------------------------------------

typedef __bf16 bf16;
typedef __attribute__((ext_vector_type(16))) __bf16 v16bf;
typedef __attribute__((ext_vector_type(8)))  __bf16 v8bf;
typedef __attribute__((ext_vector_type(8)))  float  v8f;

#define NB     64
#define TT     512
#define DD     1024
#define HH     1024
#define FOURH  4096
#define KT_D   (DD / 32)
#define KT_H   (HH / 32)

// ---- workspace layout (byte offsets, 256-aligned) ----
#define WS_PWX   0ull                                  // 8 MiB
#define WS_PWH   (WS_PWX  + 8ull * 1024 * 1024)        // 8 MiB
#define WS_XBF   (WS_PWH  + 8ull * 1024 * 1024)        // 64 MiB (x as bf16)
#define WS_HBUF  (WS_XBF  + 64ull * 1024 * 1024)       // 256 KiB
#define WS_BARR  (WS_HBUF + 256ull * 1024)             // barrier words
#define WS_XW    (WS_BARR + 256ull)                    // [T][N][4096] f32 = 512 MiB

// ---------------- fragment loaders ----------------
// A fragment (16x32 bf16) from row-major bf16 [rows][stride].
// ISA layout: lanes 0-15 row M=lane, K elems {kb..kb+7, kb+16..kb+23};
//             lanes 16-31 row M=lane-16, K elems {kb+8..kb+15, kb+24..kb+31}.
__device__ inline v16bf load_a_frag(const bf16* __restrict__ base, int rowBase,
                                    int stride, int kBase, int lane) {
  int r = rowBase + (lane & 15);
  int k = kBase + ((lane & 16) ? 8 : 0);
  const bf16* p = base + (size_t)r * stride + k;
  v8bf lo = *(const v8bf*)p;          // K = k .. k+7
  v8bf hi = *(const v8bf*)(p + 16);   // K = k+16 .. k+23
  return __builtin_shufflevector(lo, hi, 0, 1, 2, 3, 4, 5, 6, 7,
                                 8, 9, 10, 11, 12, 13, 14, 15);
}

__device__ inline v8f wmma_bf16(v16bf a, v16bf b, v8f c) {
  return __builtin_amdgcn_wmma_f32_16x16x32_bf16(
      /*neg_a=*/false, a, /*neg_b=*/false, b,
      /*c_mod=*/(short)0, c, /*reuse_a=*/false, /*reuse_b=*/false);
}

// ---------------- phase 0: prep kernels ----------------
// Pack row-major f32 W[K][NC] into bf16 WMMA B-fragments.
// Packed index = (((nt*KT + kt)*32 + lane)*16 + e)
//   -> element (k = kt*32 + (lane&16 ? 8:0) + (e&7) + (e&8 ? 16:0), n = nt*16 + (lane&15))
__global__ void pack_weights(const float* __restrict__ W, bf16* __restrict__ pW,
                             int K, int NC) {
  int kTiles = K >> 5;
  size_t total = (size_t)K * NC;
  size_t stride = (size_t)gridDim.x * blockDim.x;
  for (size_t idx = (size_t)blockIdx.x * blockDim.x + threadIdx.x; idx < total;
       idx += stride) {
    int e    = (int)(idx & 15);
    int lane = (int)((idx >> 4) & 31);
    size_t t2 = idx >> 9;
    int kt = (int)(t2 % kTiles);
    int nt = (int)(t2 / kTiles);
    int k = kt * 32 + ((lane & 16) ? 8 : 0) + (e & 7) + ((e & 8) ? 16 : 0);
    int n = nt * 16 + (lane & 15);
    pW[idx] = (bf16)W[(size_t)k * NC + n];
  }
}

__global__ void cvt_f32_bf16(const float* __restrict__ src,
                             bf16* __restrict__ dst, size_t n) {
  size_t stride = (size_t)gridDim.x * blockDim.x;
  for (size_t i = (size_t)blockIdx.x * blockDim.x + threadIdx.x; i < n;
       i += stride)
    dst[i] = (bf16)src[i];
}

__global__ void init_state(const float* __restrict__ h0,
                           bf16* __restrict__ hbuf,
                           unsigned* __restrict__ barrier_w) {
  size_t i = (size_t)blockIdx.x * blockDim.x + threadIdx.x;
  if (i < (size_t)NB * HH) hbuf[i] = (bf16)h0[i];
  if (i == 0) { barrier_w[0] = 0u; barrier_w[1] = 0u; }
}

// ---------------- phase 1: xW = x @ Wx + b ----------------
// Block: 8 waves, 128 rows x 64 cols. B slice (4 nt x 32 kt) staged in
// 128 KiB LDS once. K-loop manually unrolled x2: buffers (a0,b0)/(a1,b1)
// alternate by code position (no register copies); every load is issued
// one WMMA-group ahead of its consumer.
// Output permuted to [T][N][4096] so phase 2 reads contiguous slices.
__global__ void __launch_bounds__(256) lstm_xw_gemm(
    const bf16* __restrict__ xbf, const bf16* __restrict__ pWx,
    const float* __restrict__ bias, float* __restrict__ xW) {
  extern __shared__ char smem[];
  v16bf* ldsB = (v16bf*)smem;  // [4 nt][32 kt][32 lane] 32B fragments

  const int lane   = threadIdx.x & 31;
  const int w      = threadIdx.x >> 5;
  const int rBase  = blockIdx.x * 128 + w * 16;  // row in [0, N*T)
  const int cBase  = blockIdx.y * 64;            // col in [0, 4096)
  const int ntBase = cBase >> 4;

  // cooperative stage of B: 4*32*32 fragments, 16 per thread
  for (int i = threadIdx.x; i < 4 * 32 * 32; i += 256) {
    int l  = i & 31;
    int kt = (i >> 5) & 31;
    int nt = i >> 10;
    ldsB[i] = *(const v16bf*)(pWx +
        ((((size_t)(ntBase + nt)) * KT_D + kt) * 32 + l) * 16);
  }
  __syncthreads();

  v8f acc[4] = {};
  v16bf a0 = load_a_frag(xbf, rBase, DD, 0, lane);
  v16bf b0[4];
#pragma unroll
  for (int nt = 0; nt < 4; ++nt) b0[nt] = ldsB[(nt * 32 + 0) * 32 + lane];

  for (int kt = 0; kt < KT_D; kt += 2) {
    int k1 = kt + 1;
    int k2 = (kt + 2) & (KT_D - 1);  // wraps on final stage (harmless reload)

    v16bf a1 = load_a_frag(xbf, rBase, DD, k1 * 32, lane);
    v16bf b1[4];
#pragma unroll
    for (int nt = 0; nt < 4; ++nt) b1[nt] = ldsB[(nt * 32 + k1) * 32 + lane];
#pragma unroll
    for (int nt = 0; nt < 4; ++nt) acc[nt] = wmma_bf16(a0, b0[nt], acc[nt]);

    a0 = load_a_frag(xbf, rBase, DD, k2 * 32, lane);
#pragma unroll
    for (int nt = 0; nt < 4; ++nt) b0[nt] = ldsB[(nt * 32 + k2) * 32 + lane];
#pragma unroll
    for (int nt = 0; nt < 4; ++nt) acc[nt] = wmma_bf16(a1, b1[nt], acc[nt]);
  }

#pragma unroll
  for (int nt = 0; nt < 4; ++nt) {
    int col = cBase + nt * 16 + (lane & 15);
    float bv = bias[col];
#pragma unroll
    for (int v = 0; v < 8; ++v) {
      int r = rBase + v + ((lane & 16) ? 8 : 0);
      int n = r >> 9;         // r / T
      int t = r & (TT - 1);   // r % T
      xW[((size_t)t * NB + n) * FOURH + col] = acc[nt][v] + bv;
    }
  }
}

// ---------------- grid-wide barrier ----------------
__device__ inline void grid_barrier(unsigned* cnt, unsigned* gen) {
  __threadfence();
  __syncthreads();
  if (threadIdx.x == 0) {
    unsigned g = __hip_atomic_load(gen, __ATOMIC_ACQUIRE,
                                   __HIP_MEMORY_SCOPE_AGENT);
    unsigned arrived = __hip_atomic_fetch_add(cnt, 1u, __ATOMIC_ACQ_REL,
                                              __HIP_MEMORY_SCOPE_AGENT);
    if (arrived == gridDim.x - 1) {
      __hip_atomic_store(cnt, 0u, __ATOMIC_RELAXED, __HIP_MEMORY_SCOPE_AGENT);
      __hip_atomic_fetch_add(gen, 1u, __ATOMIC_RELEASE,
                             __HIP_MEMORY_SCOPE_AGENT);
    } else {
      while (__hip_atomic_load(gen, __ATOMIC_ACQUIRE,
                               __HIP_MEMORY_SCOPE_AGENT) == g)
        __builtin_amdgcn_s_sleep(2);
    }
  }
  __syncthreads();
}

// ---------------- phase 2: persistent recurrent scan ----------------
// 32 blocks x 8 waves = 256 waves; one 16x16 h-tile per wave.
// Block covers 2 ht tiles; waves: mt = w&3, htl = w>>2.
// All Wh fragments for the block (2 ht x 4 gates x 32 kt = 256 KiB) are
// staged in LDS ONCE and reused for all 512 steps. K-loop 2-stage
// pipelined (no copies); cell state c register-resident.
__global__ void __launch_bounds__(256) lstm_recurrent(
    const bf16* __restrict__ pWh, const float* __restrict__ xW,
    bf16* __restrict__ hbuf, float* __restrict__ out,
    unsigned* __restrict__ barrier_w) {
  extern __shared__ char smem[];
  v16bf* ldsB = (v16bf*)smem;  // [2 htl][4 gate][32 kt][32 lane]

  const int lane    = threadIdx.x & 31;
  const int w       = threadIdx.x >> 5;
  const int mt      = w & 3;               // row tile (16 rows each)
  const int htl     = w >> 2;              // local ht 0..1
  const int ht      = blockIdx.x * 2 + htl;
  const int rowBase = mt * 16;
  const int colBase = ht * 16;

  // one-time stage of Wh fragments: 2*4*32*32 = 8192 chunks, 32 per thread
  for (int i = threadIdx.x; i < 2 * 4 * 32 * 32; i += 256) {
    int l    = i & 31;
    int kt   = (i >> 5) & 31;
    int gate = (i >> 10) & 3;
    int hl   = i >> 12;
    int nt   = gate * 64 + (blockIdx.x * 2 + hl);
    ldsB[i] = *(const v16bf*)(pWh +
        (((size_t)nt * KT_H + kt) * 32 + l) * 16);
  }
  __syncthreads();
  const v16bf* myB = ldsB + (size_t)htl * 4 * 32 * 32;

  unsigned* cnt = barrier_w;
  unsigned* gen = barrier_w + 1;

  v8f c = {};  // fp32 cell state tile, register-resident across all steps

  for (int t = 0; t < TT; ++t) {
    const bf16* hb  = hbuf + (size_t)(t & 1) * NB * HH;
    bf16*       hbn = hbuf + (size_t)((t + 1) & 1) * NB * HH;

    v8f acc[4] = {};  // i, f, o, g
    v16bf a0 = load_a_frag(hb, rowBase, HH, 0, lane);
    v16bf b0[4];
#pragma unroll
    for (int g = 0; g < 4; ++g) b0[g] = myB[(g * 32 + 0) * 32 + lane];

    for (int kt = 0; kt < KT_H; kt += 2) {
      int k1 = kt + 1;
      int k2 = (kt + 2) & (KT_H - 1);  // wraps on final stage

      v16bf a1 = load_a_frag(hb, rowBase, HH, k1 * 32, lane);
      v16bf b1[4];
#pragma unroll
      for (int g = 0; g < 4; ++g) b1[g] = myB[(g * 32 + k1) * 32 + lane];
#pragma unroll
      for (int g = 0; g < 4; ++g) acc[g] = wmma_bf16(a0, b0[g], acc[g]);

      a0 = load_a_frag(hb, rowBase, HH, k2 * 32, lane);
#pragma unroll
      for (int g = 0; g < 4; ++g) b0[g] = myB[(g * 32 + k2) * 32 + lane];
#pragma unroll
      for (int g = 0; g < 4; ++g) acc[g] = wmma_bf16(a1, b1[g], acc[g]);
    }

    const float* xwt = xW + (size_t)t * NB * FOURH;  // contiguous 1MB slice
#pragma unroll
    for (int v = 0; v < 8; ++v) {
      int m    = rowBase + v + ((lane & 16) ? 8 : 0);  // batch index
      int colH = colBase + (lane & 15);
      const float* xr = xwt + (size_t)m * FOURH + colH;
      float pi = acc[0][v] + xr[0];
      float pf = acc[1][v] + xr[1024];
      float po = acc[2][v] + xr[2048];
      float pg = acc[3][v] + xr[3072];
      float ig = 1.f / (1.f + __expf(-pi));
      float fg = 1.f / (1.f + __expf(-pf));
      float og = 1.f / (1.f + __expf(-po));
      float gg = tanhf(pg);
      float cn = fg * c[v] + ig * gg;
      c[v] = cn;
      float hn = og * tanhf(cn);
      out[((size_t)m * TT + t) * HH + colH] = hn;          // [N,T,H] f32
      hbn[(size_t)m * HH + colH] = (bf16)hn;               // next-step A
    }

    // Prefetch next step's xW slice (256 waves * 32 lanes * 128B = 1 MiB).
    if (t + 1 < TT) {
      int gw = blockIdx.x * 8 + w;
      const char* nf = (const char*)(xW + (size_t)(t + 1) * NB * FOURH) +
                       ((size_t)gw * 32 + lane) * 128;
      __builtin_prefetch(nf, 0, 0);
    }

    grid_barrier(cnt, gen);  // h ping-pong: release writes, acquire next read
  }
}

// ---------------- host launch ----------------
extern "C" void kernel_launch(void* const* d_in, const int* in_sizes, int n_in,
                              void* d_out, int out_size, void* d_ws,
                              size_t ws_size, hipStream_t stream) {
  const float* x  = (const float*)d_in[0];
  const float* h0 = (const float*)d_in[1];
  const float* Wx = (const float*)d_in[2];
  const float* Wh = (const float*)d_in[3];
  const float* b  = (const float*)d_in[4];
  float* out = (float*)d_out;

  char* ws = (char*)d_ws;
  bf16* pWx       = (bf16*)(ws + WS_PWX);
  bf16* pWh       = (bf16*)(ws + WS_PWH);
  bf16* xbf       = (bf16*)(ws + WS_XBF);
  bf16* hbuf      = (bf16*)(ws + WS_HBUF);
  unsigned* barr  = (unsigned*)(ws + WS_BARR);
  float* xW       = (float*)(ws + WS_XW);

  pack_weights<<<2048, 256, 0, stream>>>(Wx, pWx, DD, FOURH);
  pack_weights<<<2048, 256, 0, stream>>>(Wh, pWh, HH, FOURH);
  cvt_f32_bf16<<<4096, 256, 0, stream>>>(x, xbf, (size_t)NB * TT * DD);
  init_state<<<(NB * HH) / 256, 256, 0, stream>>>(h0, hbuf, barr);

  dim3 g1((NB * TT) / 128, FOURH / 64);
  lstm_xw_gemm<<<g1, 256, 128 * 1024, stream>>>(xbf, pWx, b, xW);

  lstm_recurrent<<<32, 256, 256 * 1024, stream>>>(pWh, xW, hbuf, out, barr);
}
